// MatchAttention_360777253080
// MI455X (gfx1250) — matrix operand order repaired and checked by reading, not compiled, not run on hardware
//
#include <hip/hip_runtime.h>

// ---------------- problem constants ----------------
constexpr int B  = 8;
constexpr int L  = 2048;     // L1 == L2
constexpr int D  = 1024;
constexpr int BM = 64;       // attention rows per workgroup
constexpr int NT = 16;       // WMMA M/N
constexpr int KT = 32;       // WMMA K for f16
constexpr int PLD = L + 8;   // padded LDS leading dim (16B skew -> no bank dup)

typedef _Float16 f16;
typedef __attribute__((ext_vector_type(8)))  f16   v8h;
typedef __attribute__((ext_vector_type(16))) f16   v16h;
typedef __attribute__((ext_vector_type(8)))  float v8f;

// ---------------- WMMA helper ----------------
__device__ __forceinline__ v8f wmma_f16(v16h a, v16h b, v8f c) {
  // (neg_a, A, neg_b, B, c_mod, C, reuse_a, reuse_b)
  return __builtin_amdgcn_wmma_f32_16x16x32_f16(false, a, false, b, (short)0, c,
                                                false, false);
}

// A-fragment 16x32 f16 from row-major source, leading dim `ld` (elements).
// Lane (half = lane>>4, r = lane&15): VGPR0-3 hold K = 16*half+0..7,
// VGPR4-7 hold K = 16*half+8..15  -> two 16-byte loads.
__device__ __forceinline__ v16h load_a_frag(const f16* base, int ld, int k0,
                                            int half, int r) {
  const f16* p = base + (size_t)r * ld + k0 + half * 16;
  v8h lo = *(const v8h*)p;
  v8h hi = *(const v8h*)(p + 8);
  return __builtin_shufflevector(lo, hi, 0, 1, 2, 3, 4, 5, 6, 7, 8, 9, 10, 11,
                                 12, 13, 14, 15);
}

// B-fragment 32x16 where B[k][n] = X[n][k0+k] for row-major X (i.e. B = X^T):
// lane (half, n = lane&15) needs K = 16*half + 0..15 -> one contiguous 32B load.
__device__ __forceinline__ v16h load_bT_frag(const f16* Xrow_n, int k0, int half) {
  return *(const v16h*)(Xrow_n + k0 + 16 * half);
}

__device__ __forceinline__ float half16_max(float x) {
#pragma unroll
  for (int m = 1; m <= 8; m <<= 1) x = fmaxf(x, __shfl_xor(x, m));
  return x;
}
__device__ __forceinline__ float half16_sum(float x) {
#pragma unroll
  for (int m = 1; m <= 8; m <<= 1) x += __shfl_xor(x, m);
  return x;
}

// ---------------- kernel 0: convert fp32->f16 AND emit transposed copy ----
// src: [L][D] f32 (one batch selected via blockIdx.z)
// dstR: [L][D] f16 row-major, dstT: [D][L] f16 (transposed)
__global__ void __launch_bounds__(256) cvt_tr(const float* __restrict__ src,
                                              f16* __restrict__ dstR,
                                              f16* __restrict__ dstT) {
  __shared__ f16 tile[32][33];
  const int b   = blockIdx.z;
  const int c0  = blockIdx.x * 32;  // column (D) origin
  const int r0  = blockIdx.y * 32;  // row (L) origin
  const int tx  = threadIdx.x;      // 0..31
  const int ty  = threadIdx.y;      // 0..7

  const float* s = src + (size_t)b * L * D;
  f16* dr = dstR + (size_t)b * L * D;
  f16* dt = dstT + (size_t)b * D * L;

#pragma unroll
  for (int i = 0; i < 4; ++i) {
    const int r = r0 + ty + i * 8;
    const float v = s[(size_t)r * D + c0 + tx];
    const f16 h = (f16)v;
    dr[(size_t)r * D + c0 + tx] = h;      // row-major f16
    tile[ty + i * 8][tx] = h;
  }
  __syncthreads();
#pragma unroll
  for (int i = 0; i < 4; ++i) {
    const int c = c0 + ty + i * 8;        // output row in D-major layout
    dt[(size_t)c * L + r0 + tx] = tile[tx][ty + i * 8];
  }
}

// ---------------- kernel 1: stats + attended_1 ----------------
// One WG = 64 query rows of batch b. 8 waves: rowgroup = w&3 (16 rows),
// n/d interleave = w>>2. Raw f16 logits stashed in LDS during the stats sweep,
// normalized in place afterwards (no S recompute), then O1 = P @ F2 with the
// B operand read from the pre-transposed F2 (contiguous 32B fragments).
__global__ void __launch_bounds__(256) attn_pass1(const f16* __restrict__ f1b,
                                                  const f16* __restrict__ f2b,
                                                  const f16* __restrict__ f2T,
                                                  float* __restrict__ out1,
                                                  float* __restrict__ statsM,
                                                  float* __restrict__ statsL) {
  extern __shared__ char smem[];
  f16*   Pbuf = (f16*)smem;                                      // [BM][PLD]
  float* cm   = (float*)(smem + (size_t)BM * PLD * sizeof(f16)); // [8][16]
  float* cl   = cm + 8 * 16;
  float* rowM = cl + 8 * 16;                                     // [BM]
  float* rowL = rowM + BM;

  const int bid  = blockIdx.x;
  const int b    = bid / (L / BM);
  const int m0   = (bid % (L / BM)) * BM;
  const int tid  = threadIdx.x;
  const int w    = tid >> 5;
  const int lane = tid & 31;
  const int half = lane >> 4;
  const int lr   = lane & 15;
  const int rg   = w & 3;
  const int nsel = w >> 2;

  const f16* Q  = f1b + ((size_t)b * L + m0) * D;  // 64 query rows
  const f16* K  = f2b + (size_t)b * L * D;         // keys, row-major
  const f16* VT = f2T + (size_t)b * D * L;         // values, transposed [D][L]

  float mrow[8], lrow[8];
#pragma unroll
  for (int v = 0; v < 8; ++v) { mrow[v] = -3.0e38f; lrow[v] = 0.f; }

  // ---- phase 1a: S tiles (pairs), online stats, stash raw f16 logits ----
#pragma clang loop unroll(disable)
  for (int it = 0; it < 32; ++it) {
    const int nt = it * 4 + nsel * 2;  // this wave: tiles nt, nt+1
    v8f c0 = {}, c1 = {};
    __builtin_prefetch(K + (size_t)((nt + 4) * NT + lr) * D, 0, 3);
#pragma clang loop unroll_count(2)
    for (int ks = 0; ks < D / KT; ++ks) {
      v16h a  = load_a_frag(Q + (size_t)rg * 16 * D, D, ks * KT, half, lr);
      v16h b0 = load_bT_frag(K + (size_t)(nt * NT + lr) * D, ks * KT, half);
      v16h b1 = load_bT_frag(K + (size_t)((nt + 1) * NT + lr) * D, ks * KT, half);
      c0 = wmma_f16(a, b0, c0);
      c1 = wmma_f16(a, b1, c1);
    }
#pragma unroll
    for (int v = 0; v < 8; ++v) {
      const size_t row = (size_t)(rg * 16 + half * 8 + v) * PLD;
      Pbuf[row + nt * NT + lr]       = (f16)c0[v];
      Pbuf[row + (nt + 1) * NT + lr] = (f16)c1[v];
      float rmax = half16_max(fmaxf(c0[v], c1[v]));
      float mnew = fmaxf(mrow[v], rmax);
      float sexp = half16_sum(__expf(c0[v] - mnew) + __expf(c1[v] - mnew));
      lrow[v] = lrow[v] * __expf(mrow[v] - mnew) + sexp;
      mrow[v] = mnew;
    }
  }

  // ---- combine the two waves sharing each rowgroup ----
  if (lr == 0) {
#pragma unroll
    for (int v = 0; v < 8; ++v) {
      cm[w * 16 + half * 8 + v] = mrow[v];
      cl[w * 16 + half * 8 + v] = lrow[v];
    }
  }
  __syncthreads();
  if (w < 4 && lane < 16) {
    float ma = cm[w * 16 + lane], la = cl[w * 16 + lane];
    float mb = cm[(w + 4) * 16 + lane], lb = cl[(w + 4) * 16 + lane];
    float mf = fmaxf(ma, mb);
    float lf = la * __expf(ma - mf) + lb * __expf(mb - mf);
    rowM[w * 16 + lane] = mf;
    rowL[w * 16 + lane] = lf;
    statsM[(size_t)b * L + m0 + w * 16 + lane] = mf;
    statsL[(size_t)b * L + m0 + w * 16 + lane] = lf;
  }
  __syncthreads();

  float rm[8], rli[8];
#pragma unroll
  for (int v = 0; v < 8; ++v) {
    const int r = rg * 16 + half * 8 + v;
    rm[v]  = rowM[r];
    rli[v] = 1.f / rowL[r];
  }

  // ---- phase 1b: normalize logits in place (same-wave-owned tiles) ----
#pragma clang loop unroll(disable)
  for (int it = 0; it < 32; ++it) {
    const int nt = it * 4 + nsel * 2;
#pragma unroll
    for (int j = 0; j < 2; ++j)
#pragma unroll
      for (int v = 0; v < 8; ++v) {
        const size_t idx =
            (size_t)(rg * 16 + half * 8 + v) * PLD + (nt + j) * NT + lr;
        const float s = (float)Pbuf[idx];
        Pbuf[idx] = (f16)(__expf(s - rm[v]) * rli[v]);
      }
  }
  __syncthreads();

  // ---- phase 2: attended_1 = P @ F2 (K-dim = 2048 keys, B from F2^T) ----
#pragma clang loop unroll(disable)
  for (int t = 0; t < 16; ++t) {
    const int d0 = (w >> 2) * (D / 2) + t * 32;  // two d-subtiles: d0, d0+16
    v8f c0 = {}, c1 = {};
#pragma clang loop unroll_count(2)
    for (int ks = 0; ks < L / KT; ++ks) {
      v16h a  = load_a_frag(Pbuf + (size_t)rg * 16 * PLD, PLD, ks * KT, half, lr);
      v16h b0 = load_bT_frag(VT + (size_t)(d0 + lr) * L, ks * KT, half);
      v16h b1 = load_bT_frag(VT + (size_t)(d0 + 16 + lr) * L, ks * KT, half);
      c0 = wmma_f16(a, b0, c0);
      c1 = wmma_f16(a, b1, c1);
    }
#pragma unroll
    for (int v = 0; v < 8; ++v) {
      const size_t row = (size_t)b * L + m0 + rg * 16 + half * 8 + v;
      out1[row * D + d0 + lr]      = c0[v];
      out1[row * D + d0 + 16 + lr] = c1[v];
    }
  }
}

// ---------------- kernel 2: attended_2 = softmax(S)^T @ F1 ----------------
// One WG = 64 rows of F2 (n). Per-m stats are precomputed; in the S' tile the
// m-dim is the WMMA N-dim, so each lane needs one scalar stat -> single sweep.
__global__ void __launch_bounds__(256) attn_pass2(const f16* __restrict__ f1b,
                                                  const f16* __restrict__ f2b,
                                                  const f16* __restrict__ f1T,
                                                  float* __restrict__ out2,
                                                  const float* __restrict__ statsM,
                                                  const float* __restrict__ statsL) {
  extern __shared__ char smem[];
  f16* Pt = (f16*)smem;  // [BM n-rows][PLD m-cols]

  const int bid  = blockIdx.x;
  const int b    = bid / (L / BM);
  const int n0   = (bid % (L / BM)) * BM;
  const int tid  = threadIdx.x;
  const int w    = tid >> 5;
  const int lane = tid & 31;
  const int half = lane >> 4;
  const int lr   = lane & 15;
  const int rg   = w & 3;
  const int nsel = w >> 2;

  const f16* Fn = f2b + ((size_t)b * L + n0) * D;  // A rows (n)
  const f16* Fm = f1b + (size_t)b * L * D;         // B cols (m), row-major
  const f16* FT = f1T + (size_t)b * D * L;         // F1 transposed [D][L]

  // ---- phase 1: T = Fn.Fm^T (pairs), normalize by per-m stats -> LDS ----
#pragma clang loop unroll(disable)
  for (int it = 0; it < 32; ++it) {
    const int mt = it * 4 + nsel * 2;
    v8f c0 = {}, c1 = {};
    __builtin_prefetch(Fm + (size_t)((mt + 4) * NT + lr) * D, 0, 3);
#pragma clang loop unroll_count(2)
    for (int ks = 0; ks < D / KT; ++ks) {
      v16h a  = load_a_frag(Fn + (size_t)rg * 16 * D, D, ks * KT, half, lr);
      v16h b0 = load_bT_frag(Fm + (size_t)(mt * NT + lr) * D, ks * KT, half);
      v16h b1 = load_bT_frag(Fm + (size_t)((mt + 1) * NT + lr) * D, ks * KT, half);
      c0 = wmma_f16(a, b0, c0);
      c1 = wmma_f16(a, b1, c1);
    }
    const float ms0 = statsM[(size_t)b * L + mt * NT + lr];
    const float li0 = 1.f / statsL[(size_t)b * L + mt * NT + lr];
    const float ms1 = statsM[(size_t)b * L + (mt + 1) * NT + lr];
    const float li1 = 1.f / statsL[(size_t)b * L + (mt + 1) * NT + lr];
#pragma unroll
    for (int v = 0; v < 8; ++v) {
      const size_t row = (size_t)(rg * 16 + half * 8 + v) * PLD;
      Pt[row + mt * NT + lr]       = (f16)(__expf(c0[v] - ms0) * li0);
      Pt[row + (mt + 1) * NT + lr] = (f16)(__expf(c1[v] - ms1) * li1);
    }
  }
  __syncthreads();

  // ---- phase 2: attended_2 = P^T @ F1 (K-dim = 2048 queries, B from F1^T) ----
#pragma clang loop unroll(disable)
  for (int t = 0; t < 16; ++t) {
    const int d0 = (w >> 2) * (D / 2) + t * 32;
    v8f c0 = {}, c1 = {};
#pragma clang loop unroll_count(2)
    for (int ks = 0; ks < L / KT; ++ks) {
      v16h a  = load_a_frag(Pt + (size_t)rg * 16 * PLD, PLD, ks * KT, half, lr);
      v16h b0 = load_bT_frag(FT + (size_t)(d0 + lr) * L, ks * KT, half);
      v16h b1 = load_bT_frag(FT + (size_t)(d0 + 16 + lr) * L, ks * KT, half);
      c0 = wmma_f16(a, b0, c0);
      c1 = wmma_f16(a, b1, c1);
    }
#pragma unroll
    for (int v = 0; v < 8; ++v) {
      const size_t row = (size_t)b * L + n0 + rg * 16 + half * 8 + v;
      out2[row * D + d0 + lr]      = c0[v];
      out2[row * D + d0 + 16 + lr] = c1[v];
    }
  }
}

// ---------------- host entry ----------------
extern "C" void kernel_launch(void* const* d_in, const int* in_sizes, int n_in,
                              void* d_out, int out_size, void* d_ws, size_t ws_size,
                              hipStream_t stream) {
  (void)in_sizes; (void)n_in; (void)out_size; (void)ws_size;

  const float* f1 = (const float*)d_in[0];
  const float* f2 = (const float*)d_in[1];

  const size_t elems = (size_t)B * L * D;  // 16.78M

  // workspace: f1 f16 | f2 f16 | f1^T f16 | f2^T f16 | row max | row sumexp
  f16*   f1b = (f16*)d_ws;
  f16*   f2b = f1b + elems;
  f16*   f1T = f2b + elems;
  f16*   f2T = f1T + elems;
  float* sM  = (float*)(f2T + elems);
  float* sL  = sM + (size_t)B * L;

  float* out1 = (float*)d_out;  // attended_1 [B,L,D]
  float* out2 = out1 + elems;   // attended_2 [B,L,D]

  const dim3 trGrid(D / 32, L / 32, B);
  const dim3 trBlk(32, 8, 1);
  cvt_tr<<<trGrid, trBlk, 0, stream>>>(f1, f1b, f1T);
  cvt_tr<<<trGrid, trBlk, 0, stream>>>(f2, f2b, f2T);

  const size_t lds1 = (size_t)BM * PLD * sizeof(f16) +
                      (8 * 16 * 2 + BM * 2) * sizeof(float);
  const size_t lds2 = (size_t)BM * PLD * sizeof(f16);

  attn_pass1<<<B * (L / BM), 256, lds1, stream>>>(f1b, f2b, f2T, out1, sM, sL);
  attn_pass2<<<B * (L / BM), 256, lds2, stream>>>(f1b, f2b, f1T, out2, sM, sL);
}
